// ReRankTransformer_15908558864850
// MI455X (gfx1250) — matrix operand order repaired and checked by reading, not compiled, not run on hardware
//
#include <hip/hip_runtime.h>

// ---------------- problem constants ----------------
#define B_    256
#define C_    128      // lhs channels == D
#define N_    100000   // rhs table rows
#define NP_   100096   // N padded to multiple of 64
#define NID_  131072
#define KSEL_ 100
#define C2_   256      // transformer width
#define ROWS_ 25600    // B_*KSEL_
#define SPAD_ 112      // KSEL padded to multiple of 16
#define KPAD_ 128      // padded K for P x V

#define AS3 __attribute__((address_space(3)))
#define AS1 __attribute__((address_space(1)))

// ---------------- WMMA helpers (CDNA5 bf16 path) ----------------
typedef __bf16 bf16_t;
typedef __bf16 v16bf __attribute__((ext_vector_type(16)));
typedef float  v8f   __attribute__((ext_vector_type(8)));
typedef float  f32x4 __attribute__((ext_vector_type(4)));
typedef int    i32x4 __attribute__((vector_size(16)));   // matches async-builtin param type

union Frag { v16bf v; f32x4 f[2]; };

__device__ __forceinline__ v8f wmma_bf16(Frag a, Frag b, v8f c) {
  return __builtin_amdgcn_wmma_f32_16x16x32_bf16(false, a.v, false, b.v, (short)0, c,
                                                 false, false);
}
// A fragment (16x32 bf16): lanes 0-15 row=lm hold K {k0..k0+7, k0+16..k0+23},
// lanes 16-31 hold K {k0+8..15, k0+24..31}.  base = row start (k-contiguous).
__device__ __forceinline__ Frag load_frag_a(const bf16_t* base, int k0, int hf) {
  Frag fr; int e0 = k0 + hf * 8;
  fr.f[0] = *(const f32x4*)(base + e0);
  fr.f[1] = *(const f32x4*)(base + e0 + 16);
  return fr;
}
// B fragment (32x16 bf16): lanes 0-15 col=lm hold K {k0..k0+7, k0+16..k0+23},
// lanes 16-31 hold K {k0+8..15, k0+24..31}.
__device__ __forceinline__ Frag load_frag_b(const bf16_t* base, int k0, int hf) {
  Frag fr; int e0 = k0 + hf * 8;
  fr.f[0] = *(const f32x4*)(base + e0);
  fr.f[1] = *(const f32x4*)(base + e0 + 16);
  return fr;
}

// ---------------- async global->LDS staging (CDNA5 ASYNCcnt path) ----------------
// Copies `chunks` x 16B from contiguous global memory into LDS, block-cooperatively.
__device__ __forceinline__ void stage_issue(bf16_t* lds, const bf16_t* g, int chunks) {
#if __has_builtin(__builtin_amdgcn_global_load_async_to_lds_b128)
  AS3 char* lp = (AS3 char*)(size_t)lds;              // generic -> LDS offset
  AS1 char* gp = (AS1 char*)(size_t)g;
  for (int i = threadIdx.x; i < chunks; i += 256) {
    __builtin_amdgcn_global_load_async_to_lds_b128((AS1 i32x4*)(gp + (size_t)i * 16),
                                                   (AS3 i32x4*)(lp + (size_t)i * 16), 0, 0);
  }
#else
  for (int i = threadIdx.x; i < chunks; i += 256)
    ((f32x4*)lds)[i] = ((const f32x4*)g)[i];
#endif
}
__device__ __forceinline__ void stage_wait() {
#if __has_builtin(__builtin_amdgcn_global_load_async_to_lds_b128) && \
    __has_builtin(__builtin_amdgcn_s_wait_asynccnt)
  __builtin_amdgcn_s_wait_asynccnt(0);
#endif
  __syncthreads();
}

// ---------------- small helper kernels ----------------

// lhs_proj = lhs @ w_proj + b_proj ; emb_off / idg_off row offsets
__global__ void rr_lhsproj(const float* lhs, const float* w, const float* bproj,
                           const float* w_oe, const float* b_oe,
                           const float* w_oi, const float* b_oi,
                           float* proj_f, bf16_t* proj_b, float* emb_off, float* idg_off) {
  __shared__ float row[C_];
  __shared__ float red[C_];
  int b = blockIdx.x, t = threadIdx.x;
  row[t] = lhs[b * C_ + t];
  __syncthreads();
  float s = bproj[t];
  #pragma unroll 8
  for (int c = 0; c < C_; c++) s += row[c] * w[c * C_ + t];
  proj_f[b * C_ + t] = s;
  proj_b[b * C_ + t] = (bf16_t)s;
  red[t] = s * w_oe[t];
  __syncthreads();
  for (int st = 64; st > 0; st >>= 1) { if (t < st) red[t] += red[t + st]; __syncthreads(); }
  if (t == 0) emb_off[b] = red[0] + b_oe[0];
  __syncthreads();
  red[t] = s * w_oi[t];
  __syncthreads();
  for (int st = 64; st > 0; st >>= 1) { if (t < st) red[t] += red[t + st]; __syncthreads(); }
  if (t == 0) idg_off[b] = red[0] + b_oi[0];
}

// lhs_uniq[b] = lhs_proj[batch[b]]  (first B entries of sorted batch)
__global__ void rr_uniq(const int* batch, const float* proj, float* uf, bf16_t* ub) {
  int b = blockIdx.x, t = threadIdx.x;
  int bb = batch[b];
  float v = proj[bb * C_ + t];
  uf[b * C_ + t] = v;
  ub[b * C_ + t] = (bf16_t)v;
}

// transpose+convert weight [Kd,Nd] fp32 -> [Nd,Kd] bf16 (k-contiguous for B fragments)
__global__ void rr_wconv(const float* w, bf16_t* out, int Kd, int Nd) {
  int e = blockIdx.x * 256 + threadIdx.x;
  if (e < Kd * Nd) { int k = e / Nd, n = e % Nd; out[n * Kd + k] = (bf16_t)w[e]; }
}

// rhs_emb_table fp32 -> bf16, zero-padded rows up to NP_
__global__ void rr_rhsconv(const float* tab, bf16_t* out) {
  int e = blockIdx.x * 256 + threadIdx.x;   // over NP_*C_
  if (e < NP_ * C_) {
    int r = e >> 7;
    out[e] = (bf16_t)((r < N_) ? tab[e] : 0.0f);
  }
}

// ---------------- embgnn logits GEMM: [256 x 100000] = lhs_bf16 x rhs_bf16^T ----------------
__global__ __launch_bounds__(256) void rr_embgnn(const bf16_t* A, const bf16_t* BT,
                                                 const float* rowoff, float* Cout) {
  __shared__ bf16_t Bt[64 * C_];          // 16 KB tile, shared by all 8 waves
  int lane = threadIdx.x & 31, wave = threadIdx.x >> 5;
  int hf = lane >> 4, lm = lane & 15;
  int m0 = blockIdx.y * 128 + wave * 16;
  int n0 = blockIdx.x * 64;
  stage_issue(Bt, BT + (size_t)n0 * C_, (64 * C_) / 8);   // async copy B tile
  Frag afr[4];                                            // preload full A strip
  const bf16_t* Ab = A + (size_t)(m0 + lm) * C_;
  #pragma unroll
  for (int ks = 0; ks < 4; ks++) afr[ks] = load_frag_a(Ab, ks * 32, hf);
  stage_wait();
  v8f acc[4] = {};
  #pragma unroll
  for (int ks = 0; ks < 4; ks++) {
    #pragma unroll
    for (int s = 0; s < 4; s++) {
      Frag bfg = load_frag_b(&Bt[(s * 16 + lm) * C_], ks * 32, hf);
      acc[s] = wmma_bf16(afr[ks], bfg, acc[s]);
    }
  }
  #pragma unroll
  for (int s = 0; s < 4; s++) {
    int col = n0 + s * 16 + lm;
    if (col >= N_) continue;
    #pragma unroll
    for (int r = 0; r < 8; r++) {
      int row = m0 + r + hf * 8;
      Cout[(size_t)row * N_ + col] = acc[s][r] + rowoff[row];
    }
  }
}

// ---------------- idgnn logits + scatter into dense logits ----------------
__global__ void rr_posinit(int* pos) {
  int e = blockIdx.x * 256 + threadIdx.x;
  if (e < N_) pos[e] = -1;
}
__global__ void rr_posscatter(const int* batch, const int* ridx, int* pos) {
  int e = blockIdx.x * 256 + threadIdx.x;
  if (e < NID_ && batch[e] == 0) pos[ridx[e]] = e;
}
__global__ void rr_idgnn(const float* lhs, const float* gnn, const int* ridx, const int* batch,
                         const float* w_head, const float* b_head, const float* idg_off,
                         float* logits) {
  int wave = threadIdx.x >> 5, lane = threadIdx.x & 31;
  int i = blockIdx.x * 8 + wave;
  if (i >= NID_) return;
  int bt = batch[i];
  float s1 = 0.f, s2 = 0.f;
  #pragma unroll
  for (int e = lane; e < C_; e += 32) {
    float g = gnn[(size_t)i * C_ + e];
    s1 += g * w_head[e];
    s2 += g * lhs[(size_t)bt * C_ + e];
  }
  for (int m = 16; m >= 1; m >>= 1) { s1 += __shfl_xor(s1, m, 32); s2 += __shfl_xor(s2, m, 32); }
  if (lane == 0) logits[(size_t)bt * N_ + ridx[i]] = s1 + b_head[0] + s2 + idg_off[bt];
}

// ---------------- top-K per row (histogram select + small argmax sort) ----------------
#define TK_BINS 2048
#define TK_CAND 3072
__device__ __forceinline__ unsigned rr_fkey(float f) {
  unsigned u = __float_as_uint(f);
  return (u & 0x80000000u) ? ~u : (u | 0x80000000u);
}
__global__ __launch_bounds__(1024) void rr_topk(const float* logits, int* tk_ws, int* out_idx) {
  __shared__ unsigned hist[TK_BINS];
  __shared__ unsigned ckey[TK_CAND];
  __shared__ int cidx[TK_CAND];
  __shared__ int ncand;
  __shared__ int thrbin;
  __shared__ unsigned long long rk[1024];
  __shared__ int rs[1024];
  int b = blockIdx.x, t = threadIdx.x;
  const float* row = logits + (size_t)b * N_;
  for (int i = t; i < TK_BINS; i += 1024) hist[i] = 0u;
  if (t == 0) ncand = 0;
  __syncthreads();
  for (int i = t; i < N_; i += 1024) {
    if (i + 8192 < N_) __builtin_prefetch(row + i + 8192, 0, 0);   // global_prefetch
    atomicAdd(&hist[rr_fkey(row[i]) >> 21], 1u);
  }
  __syncthreads();
  if (t == 0) {
    unsigned cum = 0; int bin = TK_BINS - 1;
    for (; bin >= 0; bin--) { cum += hist[bin]; if (cum >= (unsigned)KSEL_) break; }
    thrbin = bin < 0 ? 0 : bin;
  }
  __syncthreads();
  unsigned T = (unsigned)thrbin;
  for (int i = t; i < N_; i += 1024) {
    unsigned k = rr_fkey(row[i]);
    if ((k >> 21) >= T) {
      int p = atomicAdd(&ncand, 1);
      if (p < TK_CAND) { ckey[p] = k; cidx[p] = i; }
    }
  }
  __syncthreads();
  int nc = ncand < TK_CAND ? ncand : TK_CAND;
  for (int j = 0; j < KSEL_; j++) {
    unsigned long long bestr = 0ull; int bests = -1;
    for (int i = t; i < nc; i += 1024) {
      unsigned long long r = ((unsigned long long)ckey[i] << 32) | (unsigned)(~cidx[i]);
      if (r > bestr) { bestr = r; bests = i; }
    }
    rk[t] = bestr; rs[t] = bests;
    __syncthreads();
    for (int st = 512; st > 0; st >>= 1) {
      if (t < st && rk[t + st] > rk[t]) { rk[t] = rk[t + st]; rs[t] = rs[t + st]; }
      __syncthreads();
    }
    if (t == 0) {
      int slot = rs[0];
      int col = (slot >= 0) ? cidx[slot] : 0;
      tk_ws[b * KSEL_ + j] = col;
      out_idx[b * KSEL_ + j] = col;
      if (slot >= 0) ckey[slot] = 0u;
    }
    __syncthreads();
  }
}

// ---------------- build seq x = [gathered emb | lhs_uniq] ----------------
__global__ void rr_seqbuild(const int* topk, const int* pos, const float* gnn,
                            const float* tab, const float* uniq_f,
                            float* x_f, bf16_t* x_b) {
  int rid = blockIdx.x;            // 0..ROWS_-1
  int b = rid / KSEL_;
  int t = threadIdx.x;             // 256
  int fi = topk[rid];
  float v;
  if (t < C_) {
    int q = pos[fi];
    v = (q >= 0) ? gnn[(size_t)q * C_ + t] : tab[(size_t)fi * C_ + t];
  } else {
    v = uniq_f[b * C_ + (t - C_)];
  }
  x_f[(size_t)rid * C2_ + t] = v;
  x_b[(size_t)rid * C2_ + t] = (bf16_t)v;
}

// ---------------- generic bf16 WMMA GEMM: C[M,N] = A[M,KD] x BT[N,KD]^T + bias ----------------
// BT rows are KD-contiguous (ldb == KD).  B tile staged via async-to-LDS, A strip preloaded.
template <int KD, int RELU>
__global__ __launch_bounds__(256) void rr_gemm(const bf16_t* A, int lda,
                                               const bf16_t* BT, const float* bias,
                                               float* Cf, bf16_t* Cb, int ldc) {
  __shared__ bf16_t Bt[64 * KD];
  int lane = threadIdx.x & 31, wave = threadIdx.x >> 5;
  int hf = lane >> 4, lm = lane & 15;
  int m0 = blockIdx.y * 128 + wave * 16;
  int n0 = blockIdx.x * 64;
  stage_issue(Bt, BT + (size_t)n0 * KD, (64 * KD) / 8);
  Frag afr[KD / 32];
  const bf16_t* Ab = A + (size_t)(m0 + lm) * lda;
  #pragma unroll
  for (int ks = 0; ks < KD / 32; ks++) afr[ks] = load_frag_a(Ab, ks * 32, hf);
  stage_wait();
  v8f acc[4] = {};
  #pragma unroll
  for (int ks = 0; ks < KD / 32; ks++) {
    #pragma unroll
    for (int s = 0; s < 4; s++) {
      Frag bfg = load_frag_b(&Bt[(s * 16 + lm) * KD], ks * 32, hf);
      acc[s] = wmma_bf16(afr[ks], bfg, acc[s]);
    }
  }
  #pragma unroll
  for (int s = 0; s < 4; s++) {
    int col = n0 + s * 16 + lm;
    float bv = bias ? bias[col] : 0.0f;
    #pragma unroll
    for (int r = 0; r < 8; r++) {
      int row = m0 + r + hf * 8;
      float v = acc[s][r] + bv;
      if (RELU && v < 0.0f) v = 0.0f;
      size_t o = (size_t)row * ldc + col;
      if (Cf) Cf[o] = v;
      if (Cb) Cb[o] = (bf16_t)v;
    }
  }
}

// ---------------- V transpose per batch: vT[b][d][jpad] (zero-padded j) ----------------
__global__ void rr_vtrans(const bf16_t* v, bf16_t* vT) {
  int e = blockIdx.x * 256 + threadIdx.x;   // 256*256*128
  int b = e >> 15;
  int rem = e & 32767;
  int d = rem >> 7, j = rem & 127;
  bf16_t val = (bf16_t)0.0f;
  if (j < KSEL_) val = v[((size_t)b * KSEL_ + j) * C2_ + d];
  vT[e] = val;
}

// ---------------- fused attention per batch: softmax(QK^T/16) V ----------------
__global__ __launch_bounds__(256) void rr_attn(const bf16_t* qb, const bf16_t* kb,
                                               const bf16_t* vT, bf16_t* ob) {
  __shared__ bf16_t Kt[SPAD_ * C2_];    // staged K rows (56 KB)
  __shared__ bf16_t P[SPAD_ * KPAD_];   // 112x128 bf16 probabilities (28 KB)
  int b = blockIdx.x;
  int lane = threadIdx.x & 31, wave = threadIdx.x >> 5;
  int hf = lane >> 4, lm = lane & 15;
  // async-stage the 100 valid K rows (rows 100..111 stay garbage -> masked below)
  stage_issue(Kt, kb + (size_t)b * KSEL_ * C2_, (KSEL_ * C2_) / 8);
  // zero the K-padding columns of P (112..127)
  for (int idx = threadIdx.x; idx < SPAD_ * KPAD_; idx += 256) {
    if ((idx & (KPAD_ - 1)) >= SPAD_) P[idx] = (bf16_t)0.0f;
  }
  stage_wait();
  if (wave < 7) {
    int i0 = wave * 16;
    v8f S[7] = {};
    int mrow = b * KSEL_ + i0 + lm; if (mrow > ROWS_ - 1) mrow = ROWS_ - 1;
    const bf16_t* Ab = qb + (size_t)mrow * C2_;
    for (int k0 = 0; k0 < C2_; k0 += 32) {
      Frag af = load_frag_a(Ab, k0, hf);
      #pragma unroll
      for (int n = 0; n < 7; n++) {
        Frag bfg = load_frag_b(&Kt[(n * 16 + lm) * C2_], k0, hf);
        S[n] = wmma_bf16(af, bfg, S[n]);
      }
    }
    const float scale = 1.0f / 16.0f;   // 1/sqrt(dh), dh=256
    #pragma unroll
    for (int r = 0; r < 8; r++) {
      float vals[7];
      float mx = -1e30f;
      #pragma unroll
      for (int n = 0; n < 7; n++) {
        int j = n * 16 + lm;
        float v = S[n][r] * scale;
        if (j >= KSEL_) v = -1e30f;
        vals[n] = v;
        if (v > mx) mx = v;
      }
      for (int m = 8; m >= 1; m >>= 1) { float o = __shfl_xor(mx, m, 32); if (o > mx) mx = o; }
      float sum = 0.f;
      #pragma unroll
      for (int n = 0; n < 7; n++) {
        float e = (vals[n] <= -1e29f) ? 0.0f : __expf(vals[n] - mx);
        vals[n] = e; sum += e;
      }
      for (int m = 8; m >= 1; m >>= 1) sum += __shfl_xor(sum, m, 32);
      float inv = 1.0f / sum;
      int irow = i0 + r + hf * 8;
      #pragma unroll
      for (int n = 0; n < 7; n++) P[irow * KPAD_ + n * 16 + lm] = (bf16_t)(vals[n] * inv);
    }
  }
  __syncthreads();
  if (wave < 7) {
    int i0 = wave * 16;
    const bf16_t* Ab = &P[(i0 + lm) * KPAD_];
    for (int pass = 0; pass < 2; pass++) {
      v8f O[8] = {};
      for (int k0 = 0; k0 < KPAD_; k0 += 32) {
        Frag af = load_frag_a(Ab, k0, hf);
        #pragma unroll
        for (int n = 0; n < 8; n++) {
          int d0 = (pass * 8 + n) * 16;
          Frag bfg = load_frag_b(vT + ((size_t)b * C2_ + d0 + lm) * KPAD_, k0, hf);
          O[n] = wmma_bf16(af, bfg, O[n]);
        }
      }
      #pragma unroll
      for (int n = 0; n < 8; n++) {
        int d0 = (pass * 8 + n) * 16;
        #pragma unroll
        for (int r = 0; r < 8; r++) {
          int i = i0 + r + hf * 8;
          if (i < KSEL_) ob[((size_t)b * KSEL_ + i) * C2_ + d0 + lm] = (bf16_t)O[n][r];
        }
      }
    }
  }
}

// ---------------- layernorm over residual (a + bsrc) ----------------
__global__ __launch_bounds__(256) void rr_ln(const float* a, const float* bsrc,
                                             const float* g, const float* beta,
                                             float* outf, bf16_t* outb) {
  __shared__ float red[C2_];
  int row = blockIdx.x, t = threadIdx.x;
  float x = a[(size_t)row * C2_ + t] + bsrc[(size_t)row * C2_ + t];
  red[t] = x;
  __syncthreads();
  for (int st = 128; st > 0; st >>= 1) { if (t < st) red[t] += red[t + st]; __syncthreads(); }
  float m = red[0] * (1.0f / C2_);
  __syncthreads();
  float d = x - m;
  red[t] = d * d;
  __syncthreads();
  for (int st = 128; st > 0; st >>= 1) { if (t < st) red[t] += red[t + st]; __syncthreads(); }
  float var = red[0] * (1.0f / C2_);
  float y = d * rsqrtf(var + 1e-5f) * g[t] + beta[t];
  if (outf) outf[(size_t)row * C2_ + t] = y;
  outb[(size_t)row * C2_ + t] = (bf16_t)y;
}

// ---------------- tr_logits = <lhs_uniq[b], seq_out[row]> ----------------
__global__ void rr_trdot(const float* uniq, const float* sout, float* out) {
  int wave = threadIdx.x >> 5, lane = threadIdx.x & 31;
  int rid = blockIdx.x * 8 + wave;
  if (rid >= ROWS_) return;
  int b = rid / KSEL_;
  float s = 0.f;
  #pragma unroll
  for (int e = lane; e < C_; e += 32) s += uniq[b * C_ + e] * sout[(size_t)rid * C_ + e];
  for (int m = 16; m >= 1; m >>= 1) s += __shfl_xor(s, m, 32);
  if (lane == 0) out[rid] = s;
}

// ---------------- host launcher ----------------
extern "C" void kernel_launch(void* const* d_in, const int* in_sizes, int n_in,
                              void* d_out, int out_size, void* d_ws, size_t ws_size,
                              hipStream_t stream) {
  (void)in_sizes; (void)n_in; (void)out_size; (void)ws_size;
  const float* lhs_emb   = (const float*)d_in[0];
  const float* rhs_gnn   = (const float*)d_in[1];
  const int*   r_index   = (const int*)d_in[2];
  const int*   l_batch   = (const int*)d_in[3];
  const float* emb_tab   = (const float*)d_in[4];
  const float* w_proj    = (const float*)d_in[5];
  const float* b_proj    = (const float*)d_in[6];
  const float* w_head    = (const float*)d_in[7];
  const float* b_head    = (const float*)d_in[8];
  const float* w_off_emb = (const float*)d_in[9];
  const float* b_off_emb = (const float*)d_in[10];
  const float* w_off_idg = (const float*)d_in[11];
  const float* b_off_idg = (const float*)d_in[12];
  const float* wq = (const float*)d_in[13];
  const float* bq = (const float*)d_in[14];
  const float* wk = (const float*)d_in[15];
  const float* bk = (const float*)d_in[16];
  const float* wv = (const float*)d_in[17];
  const float* bv = (const float*)d_in[18];
  const float* wo = (const float*)d_in[19];
  const float* bo = (const float*)d_in[20];
  const float* ln1_g = (const float*)d_in[21];
  const float* ln1_b = (const float*)d_in[22];
  const float* wf  = (const float*)d_in[23];
  const float* bfc = (const float*)d_in[24];
  const float* ln2_g = (const float*)d_in[25];
  const float* ln2_b = (const float*)d_in[26];
  const float* w_tr = (const float*)d_in[27];
  const float* b_tr = (const float*)d_in[28];

  float* out_logits = (float*)d_out;
  float* out_tr     = out_logits + (size_t)B_ * N_;
  int*   out_idx    = (int*)(out_logits + (size_t)B_ * N_ + (size_t)B_ * KSEL_);

  // ---- workspace carve-out ----
  char* wsb = (char*)d_ws;
  size_t off = 0;
  auto alloc = [&](size_t bytes) -> void* {
    void* p = wsb + off;
    off = (off + bytes + 255) & ~(size_t)255;
    return p;
  };
  float*  proj_f = (float*)alloc((size_t)B_ * C_ * 4);
  bf16_t* proj_b = (bf16_t*)alloc((size_t)B_ * C_ * 2);
  float*  uniq_f = (float*)alloc((size_t)B_ * C_ * 4);
  bf16_t* uniq_b = (bf16_t*)alloc((size_t)B_ * C_ * 2);
  float*  emb_off = (float*)alloc(B_ * 4);
  float*  idg_off = (float*)alloc(B_ * 4);
  int*    tk_ws   = (int*)alloc((size_t)B_ * KSEL_ * 4);
  int*    pos_s   = (int*)alloc((size_t)N_ * 4);
  bf16_t* wqT  = (bf16_t*)alloc((size_t)C2_ * C2_ * 2);
  bf16_t* wkT  = (bf16_t*)alloc((size_t)C2_ * C2_ * 2);
  bf16_t* wvT  = (bf16_t*)alloc((size_t)C2_ * C2_ * 2);
  bf16_t* woT  = (bf16_t*)alloc((size_t)C2_ * C2_ * 2);
  bf16_t* wfT  = (bf16_t*)alloc((size_t)C2_ * C2_ * 2);
  bf16_t* wtrT = (bf16_t*)alloc((size_t)C_ * C2_ * 2);
  float*  x_f  = (float*)alloc((size_t)ROWS_ * C2_ * 4);
  bf16_t* x_b  = (bf16_t*)alloc((size_t)ROWS_ * C2_ * 2);
  bf16_t* q_b  = (bf16_t*)alloc((size_t)ROWS_ * C2_ * 2);
  bf16_t* k_b  = (bf16_t*)alloc((size_t)ROWS_ * C2_ * 2);
  bf16_t* v_b  = (bf16_t*)alloc((size_t)ROWS_ * C2_ * 2);
  bf16_t* vT   = (bf16_t*)alloc((size_t)B_ * C2_ * KPAD_ * 2);
  bf16_t* o_b  = (bf16_t*)alloc((size_t)ROWS_ * C2_ * 2);
  float*  tmp_f = (float*)alloc((size_t)ROWS_ * C2_ * 4);
  float*  h_f  = (float*)alloc((size_t)ROWS_ * C2_ * 4);
  bf16_t* h_b  = (bf16_t*)alloc((size_t)ROWS_ * C2_ * 2);
  bf16_t* y_b  = (bf16_t*)alloc((size_t)ROWS_ * C2_ * 2);
  float*  s_out = (float*)alloc((size_t)ROWS_ * C_ * 4);
  // rhs bf16 table aliases q_b+k_b (dead before q/k are produced): 25.6MB <= 26.2MB
  bf16_t* rhs_b = q_b;

  // ---- phase 0: projections / conversions ----
  rr_lhsproj<<<B_, C_, 0, stream>>>(lhs_emb, w_proj, b_proj, w_off_emb, b_off_emb,
                                    w_off_idg, b_off_idg, proj_f, proj_b, emb_off, idg_off);
  rr_uniq<<<B_, C_, 0, stream>>>(l_batch, proj_f, uniq_f, uniq_b);
  rr_wconv<<<256, 256, 0, stream>>>(wq, wqT, C2_, C2_);
  rr_wconv<<<256, 256, 0, stream>>>(wk, wkT, C2_, C2_);
  rr_wconv<<<256, 256, 0, stream>>>(wv, wvT, C2_, C2_);
  rr_wconv<<<256, 256, 0, stream>>>(wo, woT, C2_, C2_);
  rr_wconv<<<256, 256, 0, stream>>>(wf, wfT, C2_, C2_);
  rr_wconv<<<128, 256, 0, stream>>>(w_tr, wtrT, C2_, C_);
  rr_rhsconv<<<(NP_ * C_ + 255) / 256, 256, 0, stream>>>(emb_tab, rhs_b);

  // ---- phase 1: dense retrieval logits (WMMA) + idgnn scatter ----
  rr_embgnn<<<dim3(NP_ / 64, 2), 256, 0, stream>>>(proj_b, rhs_b, emb_off, out_logits);
  rr_posinit<<<(N_ + 255) / 256, 256, 0, stream>>>(pos_s);
  rr_posscatter<<<NID_ / 256, 256, 0, stream>>>(l_batch, r_index, pos_s);
  rr_idgnn<<<NID_ / 8, 256, 0, stream>>>(lhs_emb, rhs_gnn, r_index, l_batch,
                                         w_head, b_head, idg_off, out_logits);

  // ---- phase 2: top-K + sequence build ----
  rr_topk<<<B_, 1024, 0, stream>>>(out_logits, tk_ws, out_idx);
  rr_seqbuild<<<ROWS_, C2_, 0, stream>>>(tk_ws, pos_s, rhs_gnn, emb_tab, uniq_f, x_f, x_b);

  // ---- phase 3: MAB (set transformer block), all GEMMs via WMMA ----
  dim3 g256(C2_ / 64, ROWS_ / 128);
  rr_gemm<256, 0><<<g256, 256, 0, stream>>>(x_b, C2_, wqT, bq, nullptr, q_b, C2_);
  rr_gemm<256, 0><<<g256, 256, 0, stream>>>(x_b, C2_, wkT, bk, nullptr, k_b, C2_);
  rr_gemm<256, 0><<<g256, 256, 0, stream>>>(x_b, C2_, wvT, bv, nullptr, v_b, C2_);
  rr_vtrans<<<(B_ * C2_ * KPAD_) / 256, 256, 0, stream>>>(v_b, vT);
  rr_attn<<<B_, 256, 0, stream>>>(q_b, k_b, vT, o_b);
  rr_gemm<256, 0><<<g256, 256, 0, stream>>>(o_b, C2_, woT, bo, tmp_f, nullptr, C2_);
  rr_ln<<<ROWS_, C2_, 0, stream>>>(x_f, tmp_f, ln1_g, ln1_b, h_f, h_b);
  rr_gemm<256, 1><<<g256, 256, 0, stream>>>(h_b, C2_, wfT, bfc, tmp_f, nullptr, C2_);
  rr_ln<<<ROWS_, C2_, 0, stream>>>(h_f, tmp_f, ln2_g, ln2_b, nullptr, y_b);
  rr_gemm<256, 0><<<dim3(C_ / 64, ROWS_ / 128), 256, 0, stream>>>(y_b, C2_, wtrT, b_tr,
                                                                  s_out, nullptr, C_);
  rr_trdot<<<ROWS_ / 8, 256, 0, stream>>>(uniq_f, s_out, out_tr);
}
